// SigLIPContrastiveLoss_40776419508314
// MI455X (gfx1250) — matrix operand order repaired and checked by reading, not compiled, not run on hardware
//
#include <hip/hip_runtime.h>
#include <math.h>

#define N_ROWS 16384
#define DIM    512
#define INV_TEMP 14.285714285714286f  /* 1/0.07 */
#define MARGIN   0.1f
#define BK 32
#define LDS_STRIDE 40   /* 32 + 8 bf16 pad to spread LDS banks */

typedef __bf16 bf16_t;
typedef __attribute__((ext_vector_type(16))) bf16_t v16bf;
typedef __attribute__((ext_vector_type(8)))  bf16_t v8bf;
typedef __attribute__((ext_vector_type(8)))  float  v8f;
typedef __attribute__((ext_vector_type(4)))  float  v4f;

__device__ __forceinline__ v16bf cat16(v8bf a, v8bf b) {
  return __builtin_shufflevector(a, b, 0,1,2,3,4,5,6,7,8,9,10,11,12,13,14,15);
}

// ---------------------------------------------------------------------------
// Kernel 1: L2-normalize each row, split into bf16 hi + lo (bf16x3 scheme).
// One wave (32 lanes) per row; each lane handles 16 consecutive floats.
// ---------------------------------------------------------------------------
__global__ void norm_split_kernel(const float* __restrict__ s,
                                  const float* __restrict__ t,
                                  bf16_t* __restrict__ shi, bf16_t* __restrict__ slo,
                                  bf16_t* __restrict__ thi, bf16_t* __restrict__ tlo) {
  const int wave = threadIdx.x >> 5;
  const int lane = threadIdx.x & 31;
  const int row  = blockIdx.x * 8 + wave;

  const float* src = (blockIdx.y == 0) ? s   : t;
  bf16_t* dhi      = (blockIdx.y == 0) ? shi : thi;
  bf16_t* dlo      = (blockIdx.y == 0) ? slo : tlo;

  const size_t base = (size_t)row * DIM + (size_t)lane * 16;
  const v4f* p = (const v4f*)(src + base);

  v4f xs[4];
  float ss = 0.0f;
#pragma unroll
  for (int j = 0; j < 4; ++j) {
    xs[j] = p[j];
#pragma unroll
    for (int c = 0; c < 4; ++c) ss += xs[j][c] * xs[j][c];
  }
#pragma unroll
  for (int off = 16; off >= 1; off >>= 1) ss += __shfl_xor(ss, off, 32);

  const float scale = 1.0f / fmaxf(sqrtf(ss), 1e-12f);

#pragma unroll
  for (int j = 0; j < 4; ++j) {
#pragma unroll
    for (int c = 0; c < 4; ++c) {
      float v = xs[j][c] * scale;
      bf16_t h = (bf16_t)v;
      bf16_t l = (bf16_t)(v - (float)h);
      dhi[base + j * 4 + c] = h;
      dlo[base + j * 4 + c] = l;
    }
  }
}

// ---------------------------------------------------------------------------
// Kernel 2: tiled bf16x3 WMMA GEMM fused with SigLIP log-sigmoid loss.
// Block = 256 threads (8 waves), C-tile 128x128; wave tile 64x32 (4x2 WMMA).
// ---------------------------------------------------------------------------
__global__ void __launch_bounds__(256)
gemm_loss_kernel(const bf16_t* __restrict__ shi, const bf16_t* __restrict__ slo,
                 const bf16_t* __restrict__ thi, const bf16_t* __restrict__ tlo,
                 float* __restrict__ partials) {
  __shared__ __align__(16) bf16_t sAhi[128 * LDS_STRIDE];
  __shared__ __align__(16) bf16_t sAlo[128 * LDS_STRIDE];
  __shared__ __align__(16) bf16_t sBhi[128 * LDS_STRIDE];
  __shared__ __align__(16) bf16_t sBlo[128 * LDS_STRIDE];
  __shared__ float red[8];

  const int tid   = threadIdx.x;
  const int lane  = tid & 31;
  const int wave  = tid >> 5;
  const int waveM = wave >> 2;   // 0..1  (64-row slabs)
  const int waveN = wave & 3;    // 0..3  (32-col slabs)
  const int m0 = blockIdx.y * 128;
  const int n0 = blockIdx.x * 128;

  // global->LDS staging: each thread moves 16 bf16 (two 128b loads) per array
  const int lrow = tid >> 1;         // 0..127
  const int lcol = (tid & 1) * 16;   // 0 or 16
  const size_t aoff = (size_t)(m0 + lrow) * DIM + lcol;
  const size_t boff = (size_t)(n0 + lrow) * DIM + lcol;

  v8f acc[4][2] = {};

  const int mrow    = lane & 15;
  const int hi_half = lane >> 4;          // lane group selects K-halves
  const int ka = hi_half ? 8 : 0;         // A chunks at {ka, ka+16}
  const int kb = hi_half ? 8 : 0;         // B chunks at {kb, kb+16}

  for (int k0 = 0; k0 < DIM; k0 += BK) {
    const uint4* gah = (const uint4*)(shi + aoff + k0);
    const uint4* gal = (const uint4*)(slo + aoff + k0);
    const uint4* gbh = (const uint4*)(thi + boff + k0);
    const uint4* gbl = (const uint4*)(tlo + boff + k0);
    uint4 ah0 = gah[0], ah1 = gah[1];
    uint4 al0 = gal[0], al1 = gal[1];
    uint4 bh0 = gbh[0], bh1 = gbh[1];
    uint4 bl0 = gbl[0], bl1 = gbl[1];

    __syncthreads();  // previous iteration's consumers are done
    {
      const int d = lrow * LDS_STRIDE + lcol;
      *(uint4*)(&sAhi[d]) = ah0;  *(uint4*)(&sAhi[d + 8]) = ah1;
      *(uint4*)(&sAlo[d]) = al0;  *(uint4*)(&sAlo[d + 8]) = al1;
      *(uint4*)(&sBhi[d]) = bh0;  *(uint4*)(&sBhi[d + 8]) = bh1;
      *(uint4*)(&sBlo[d]) = bl0;  *(uint4*)(&sBlo[d + 8]) = bl1;
    }
    __syncthreads();

    // A fragments (16x32 bf16): lanes 0-15 hold K{0..7,16..23}, lanes 16-31 K{8..15,24..31}
    v16bf afh[4], afl[4];
#pragma unroll
    for (int mi = 0; mi < 4; ++mi) {
      const int r = (waveM * 64 + mi * 16 + mrow) * LDS_STRIDE;
      afh[mi] = cat16(*(const v8bf*)(&sAhi[r + ka]), *(const v8bf*)(&sAhi[r + ka + 16]));
      afl[mi] = cat16(*(const v8bf*)(&sAlo[r + ka]), *(const v8bf*)(&sAlo[r + ka + 16]));
    }
    // B fragments (32x16 bf16): lanes 0-15 hold K{0..7,16..23} of col `lane`; lanes 16-31 K{8..15,24..31}
    v16bf bfh[2], bfl[2];
#pragma unroll
    for (int ni = 0; ni < 2; ++ni) {
      const int r = (waveN * 32 + ni * 16 + mrow) * LDS_STRIDE;
      bfh[ni] = cat16(*(const v8bf*)(&sBhi[r + kb]), *(const v8bf*)(&sBhi[r + kb + 16]));
      bfl[ni] = cat16(*(const v8bf*)(&sBlo[r + kb]), *(const v8bf*)(&sBlo[r + kb + 16]));
    }

#pragma unroll
    for (int mi = 0; mi < 4; ++mi)
#pragma unroll
      for (int ni = 0; ni < 2; ++ni) {
        // bf16x3: hi*hi + hi*lo + lo*hi  (fp32-grade dot product)
        acc[mi][ni] = __builtin_amdgcn_wmma_f32_16x16x32_bf16(
            false, afh[mi], false, bfh[ni], (short)0, acc[mi][ni], false, false);
        acc[mi][ni] = __builtin_amdgcn_wmma_f32_16x16x32_bf16(
            false, afh[mi], false, bfl[ni], (short)0, acc[mi][ni], false, false);
        acc[mi][ni] = __builtin_amdgcn_wmma_f32_16x16x32_bf16(
            false, afl[mi], false, bfh[ni], (short)0, acc[mi][ni], false, false);
      }
  }

  // Epilogue: logits -> log_sigmoid(target*logit + margin), accumulate.
  // C layout (f32 16x16): VGPR r holds M = r + 8*(lane>=16); N = lane%16.
  float local = 0.0f;
#pragma unroll
  for (int mi = 0; mi < 4; ++mi) {
#pragma unroll
    for (int ni = 0; ni < 2; ++ni) {
      const int gm_base = m0 + waveM * 64 + mi * 16 + hi_half * 8;
      const int gn      = n0 + waveN * 32 + ni * 16 + mrow;
#pragma unroll
      for (int r = 0; r < 8; ++r) {
        const int gm = gm_base + r;
        const float logit = acc[mi][ni][r] * INV_TEMP;
        const float tgt = (gm == gn) ? 1.0f : -1.0f;
        const float x = tgt * logit + MARGIN;
        // log_sigmoid(x) = min(x,0) - log1p(exp(-|x|))
        local += fminf(x, 0.0f) - log1pf(expf(-fabsf(x)));
      }
    }
  }

#pragma unroll
  for (int off = 16; off >= 1; off >>= 1) local += __shfl_xor(local, off, 32);
  if (lane == 0) red[wave] = local;
  __syncthreads();
  if (tid == 0) {
    float ssum = 0.0f;
#pragma unroll
    for (int w = 0; w < 8; ++w) ssum += red[w];
    partials[blockIdx.y * gridDim.x + blockIdx.x] = ssum;
  }
}

// ---------------------------------------------------------------------------
// Kernel 3: deterministic fixed-order reduction of 16384 block partials.
// ---------------------------------------------------------------------------
__global__ void final_reduce_kernel(const float* __restrict__ partials,
                                    float* __restrict__ out) {
  __shared__ float sm[256];
  float ssum = 0.0f;
  for (int i = threadIdx.x; i < 16384; i += 256) ssum += partials[i];
  sm[threadIdx.x] = ssum;
  __syncthreads();
  for (int off = 128; off >= 1; off >>= 1) {
    if ((int)threadIdx.x < off) sm[threadIdx.x] += sm[threadIdx.x + off];
    __syncthreads();
  }
  if (threadIdx.x == 0) out[0] = -sm[0] / (float)N_ROWS;
}

// ---------------------------------------------------------------------------
extern "C" void kernel_launch(void* const* d_in, const int* in_sizes, int n_in,
                              void* d_out, int out_size, void* d_ws, size_t ws_size,
                              hipStream_t stream) {
  (void)in_sizes; (void)n_in; (void)out_size; (void)ws_size;

  const float* s = (const float*)d_in[0];
  const float* t = (const float*)d_in[1];
  const size_t ND = (size_t)N_ROWS * DIM;

  bf16_t* shi = (bf16_t*)d_ws;
  bf16_t* slo = shi + ND;
  bf16_t* thi = slo + ND;
  bf16_t* tlo = thi + ND;
  float* partials = (float*)((char*)d_ws + 4 * ND * sizeof(bf16_t));  // +64 MB

  norm_split_kernel<<<dim3(N_ROWS / 8, 2), 256, 0, stream>>>(s, t, shi, slo, thi, tlo);
  gemm_loss_kernel<<<dim3(128, 128), 256, 0, stream>>>(shi, slo, thi, tlo, partials);
  final_reduce_kernel<<<1, 256, 0, stream>>>(partials, (float*)d_out);
}
